// multiHead_att_44006234915476
// MI455X (gfx1250) — matrix operand order repaired and checked by reading, not compiled, not run on hardware
//
#include <hip/hip_runtime.h>

// ---------------------------------------------------------------------------
// MI455X (gfx1250) attention: X@Wq/Wk/Wv -> QK^T (masked, scaled) -> softmax
// -> P@V.  bf16 operands, fp32 WMMA accumulation (v_wmma_f32_16x16x32_bf16).
// All GEMMs are NT (B stored [N][K]); tiles staged to LDS with async
// global->LDS copies (ASYNCcnt) and double buffering when available.
// ---------------------------------------------------------------------------

typedef __attribute__((ext_vector_type(16))) __bf16 v16bf;
typedef __attribute__((ext_vector_type(8)))  float  v8f;
typedef __attribute__((ext_vector_type(4)))  int    v4i;

struct __attribute__((aligned(16))) U4 { unsigned int x, y, z, w; };
struct __attribute__((aligned(16))) U8 { U4 lo, hi; };
static_assert(sizeof(U8) == 32, "U8 must be 32 bytes");

#define BATCH_N 8
#define SEQ_T   2048
#define DIM_E   1024
#define DIM_H   1024
#define BK      32
#define LDP     40   // padded LDS row pitch in ushorts (80B, 16B aligned)

// ---- async global->LDS copy support (guarded; falls back to sync copy) ----
#if defined(__has_builtin)
#if __has_builtin(__builtin_amdgcn_global_load_async_to_lds_b128)
#define USE_ASYNC_LDS 1
#endif
#endif
#ifndef USE_ASYNC_LDS
#define USE_ASYNC_LDS 0
#endif

__device__ __forceinline__ void cp_b128(const unsigned short* g, unsigned short* l) {
#if USE_ASYNC_LDS
    typedef __attribute__((address_space(1))) v4i* gp_t;
    typedef __attribute__((address_space(3))) v4i* lp_t;
    __builtin_amdgcn_global_load_async_to_lds_b128(
        (gp_t)(unsigned long long)(__SIZE_TYPE__)g,
        (lp_t)(unsigned int)(__SIZE_TYPE__)l, 0, 0);
#else
    *(U4*)l = *(const U4*)g;
#endif
}

template <int N>
__device__ __forceinline__ void wait_async() {
#if USE_ASYNC_LDS
#if __has_builtin(__builtin_amdgcn_s_wait_asynccnt)
    __builtin_amdgcn_s_wait_asynccnt(N);
#else
    asm volatile("s_wait_asynccnt %0" :: "i"(N) : "memory");
#endif
#endif
}

__device__ __forceinline__ unsigned short f32_to_bf16(float f) {
    unsigned int u = __builtin_bit_cast(unsigned int, f);
    unsigned int r = (u + 0x7FFFu + ((u >> 16) & 1u)) >> 16;
    return (unsigned short)r;
}
__device__ __forceinline__ float bf16_to_f32(unsigned short s) {
    unsigned int u = ((unsigned int)s) << 16;
    return __builtin_bit_cast(float, u);
}
__device__ __forceinline__ v16bf load_frag(const unsigned short* p0,
                                           const unsigned short* p1) {
    U8 t;
    t.lo = *(const U4*)p0;
    t.hi = *(const U4*)p1;
    return __builtin_bit_cast(v16bf, t);
}

// --------------------------- fp32 -> bf16 convert ---------------------------
__global__ void cvt_f32_bf16_kernel(const float* __restrict__ src,
                                    unsigned short* __restrict__ dst, long n4) {
    long i = (long)blockIdx.x * blockDim.x + threadIdx.x;
    long stride = (long)gridDim.x * blockDim.x;
    for (; i < n4; i += stride) {
        float4 f = *(const float4*)(src + i * 4);
        *(unsigned int*)(dst + i * 4) =
            (unsigned int)f32_to_bf16(f.x) | ((unsigned int)f32_to_bf16(f.y) << 16);
        *(unsigned int*)(dst + i * 4 + 2) =
            (unsigned int)f32_to_bf16(f.z) | ((unsigned int)f32_to_bf16(f.w) << 16);
    }
}

// ---------------- fp32 [E][H] -> bf16 transposed [H][E] --------------------
__global__ void cvt_transpose_kernel(const float* __restrict__ src,
                                     unsigned short* __restrict__ dst) {
    long i = (long)blockIdx.x * blockDim.x + threadIdx.x;
    long stride = (long)gridDim.x * blockDim.x;
    const long n = (long)DIM_E * DIM_H;
    for (; i < n; i += stride) {
        long h = i >> 10;            // DIM_E == 1024
        long e = i & (DIM_E - 1);
        dst[h * DIM_E + e] = f32_to_bf16(src[e * DIM_H + h]);   // coalesced writes
    }
}

// ----------------------------- unified NT GEMM -----------------------------
// C = A[M,K] @ B[N,K]^T, bf16 operands, fp32 accumulation.
// EPI: 0 = bf16 row-major out, 1 = bf16 per-batch transposed out (V),
//      2 = fp32 row-major out,  3 = energy (mask + scale, fp32 out)
#define EPI_BF16   0
#define EPI_BF16_T 1
#define EPI_F32    2
#define EPI_ENERGY 3

template <int EPI>
__global__ __launch_bounds__(256)
void gemm_nt_kernel(const unsigned short* __restrict__ A,
                    const unsigned short* __restrict__ B,
                    void* __restrict__ Cv,
                    int M, int N, int K,
                    long strideA, long strideB, long strideC,
                    const int* __restrict__ lengths,
                    const int* __restrict__ dk_ptr) {
    __shared__ unsigned short As[2][128 * LDP];
    __shared__ unsigned short Bs[2][128 * LDP];

    A += (size_t)blockIdx.z * strideA;
    B += (size_t)blockIdx.z * strideB;
    float* Cf          = (float*)Cv          + (size_t)blockIdx.z * strideC;
    unsigned short* Cb = (unsigned short*)Cv + (size_t)blockIdx.z * strideC;

    const int tid = threadIdx.x;
    const int m0 = blockIdx.y * 128;
    const int n0 = blockIdx.x * 128;
    const float NEG_INF = -__builtin_inff();

    // energy: tile entirely above the diagonal -> fully masked, skip matmul
    if (EPI == EPI_ENERGY && n0 > m0 + 127) {
        for (int i = tid; i < 128 * 128; i += 256) {
            int r = i >> 7, c = i & 127;
            Cf[(size_t)(m0 + r) * N + n0 + c] = NEG_INF;
        }
        return;
    }

    const int lane  = tid & 31;
    const int wave  = tid >> 5;
    const int waveM = wave & 3;      // 4 waves along M (32 rows each)
    const int waveN = wave >> 2;     // 2 waves along N (64 cols each)
    const int halfK = lane >> 4;
    const int l16   = lane & 15;

    const unsigned short* Atile = A + (size_t)m0 * K;
    const unsigned short* Btile = B + (size_t)n0 * K;
    const int stR  = tid >> 2;        // staging row   (0..127 over 2 iters)
    const int stS  = tid & 3;         // staging K-seg (0..3)

    v8f acc[2][4] = {};

    const int nk = K / BK;
    // prologue: stage k-tile 0 into buffer 0 (4 x b128 per thread)
    #pragma unroll
    for (int i = 0; i < 2; ++i) {
        int r = stR + i * 64;
        cp_b128(Atile + (size_t)r * K + stS * 8, &As[0][r * LDP + stS * 8]);
        cp_b128(Btile + (size_t)r * K + stS * 8, &Bs[0][r * LDP + stS * 8]);
    }

    int buf = 0;
    for (int t = 0; t < nk; ++t) {
        const int k0n = (t + 1) * BK;
        if (t + 1 < nk) {
            #pragma unroll
            for (int i = 0; i < 2; ++i) {
                int r = stR + i * 64;
                cp_b128(Atile + (size_t)r * K + k0n + stS * 8,
                        &As[buf ^ 1][r * LDP + stS * 8]);
                cp_b128(Btile + (size_t)r * K + k0n + stS * 8,
                        &Bs[buf ^ 1][r * LDP + stS * 8]);
            }
            wait_async<4>();     // stage t complete; stage t+1 in flight
        } else {
            wait_async<0>();
        }
        __syncthreads();

        v16bf a[2], b[4];
        #pragma unroll
        for (int tm = 0; tm < 2; ++tm) {
            const unsigned short* base =
                &As[buf][(waveM * 32 + tm * 16 + l16) * LDP + halfK * 8];
            a[tm] = load_frag(base, base + 16);
        }
        #pragma unroll
        for (int tn = 0; tn < 4; ++tn) {
            const unsigned short* base =
                &Bs[buf][(waveN * 64 + tn * 16 + l16) * LDP + halfK * 8];
            b[tn] = load_frag(base, base + 16);
        }
        #pragma unroll
        for (int tm = 0; tm < 2; ++tm)
            #pragma unroll
            for (int tn = 0; tn < 4; ++tn)
                acc[tm][tn] = __builtin_amdgcn_wmma_f32_16x16x32_bf16(
                    false, a[tm], false, b[tn], (short)0, acc[tm][tn],
                    false, false);
        __syncthreads();          // all waves done reading buf before refill
        buf ^= 1;
    }

    // ------------------------------ epilogue -------------------------------
    int   len   = 0;
    float scale = 1.0f;
    if (EPI == EPI_ENERGY) {
        len   = lengths[blockIdx.z];
        scale = rsqrtf((float)dk_ptr[0]);
    }

    #pragma unroll
    for (int tm = 0; tm < 2; ++tm)
        #pragma unroll
        for (int tn = 0; tn < 4; ++tn) {
            int mb = m0 + waveM * 32 + tm * 16 + halfK * 8;
            int nb = n0 + waveN * 64 + tn * 16 + l16;
            if (EPI == EPI_BF16) {
                #pragma unroll
                for (int r = 0; r < 8; ++r)
                    Cb[(size_t)(mb + r) * N + nb] = f32_to_bf16(acc[tm][tn][r]);
            } else if (EPI == EPI_BF16_T) {
                // C layout: [batch][N][SEQ_T]; mb -> (batch, t); 8 contiguous t
                int bI = mb >> 11;                // SEQ_T == 2048
                int t0 = mb & (SEQ_T - 1);
                U4 o;
                o.x = (unsigned int)f32_to_bf16(acc[tm][tn][0]) |
                      ((unsigned int)f32_to_bf16(acc[tm][tn][1]) << 16);
                o.y = (unsigned int)f32_to_bf16(acc[tm][tn][2]) |
                      ((unsigned int)f32_to_bf16(acc[tm][tn][3]) << 16);
                o.z = (unsigned int)f32_to_bf16(acc[tm][tn][4]) |
                      ((unsigned int)f32_to_bf16(acc[tm][tn][5]) << 16);
                o.w = (unsigned int)f32_to_bf16(acc[tm][tn][6]) |
                      ((unsigned int)f32_to_bf16(acc[tm][tn][7]) << 16);
                *(U4*)(Cb + (size_t)bI * N * SEQ_T + (size_t)nb * SEQ_T + t0) = o;
            } else if (EPI == EPI_F32) {
                #pragma unroll
                for (int r = 0; r < 8; ++r)
                    Cf[(size_t)(mb + r) * N + nb] = acc[tm][tn][r];
            } else { // EPI_ENERGY
                #pragma unroll
                for (int r = 0; r < 8; ++r) {
                    int i = mb + r;
                    bool masked = (nb > i) || (nb >= len);
                    Cf[(size_t)i * N + nb] = masked ? NEG_INF
                                                    : acc[tm][tn][r] * scale;
                }
            }
        }
}

// ------------------------------ lengths[b] ---------------------------------
__global__ void lengths_kernel(const unsigned short* __restrict__ Q,
                               const unsigned short* __restrict__ Km,
                               int* __restrict__ lengths) {
    const int b = blockIdx.x;
    const unsigned short* q0 = Q  + (size_t)b * SEQ_T * DIM_H;
    const unsigned short* Kb = Km + (size_t)b * SEQ_T * DIM_H;
    __shared__ int cnt;
    if (threadIdx.x == 0) cnt = 0;
    __syncthreads();
    int local = 0;
    for (int j = threadIdx.x; j < SEQ_T; j += blockDim.x) {
        const unsigned short* kr = Kb + (size_t)j * DIM_H;
        float d = 0.f;
        for (int e = 0; e < DIM_H; ++e)
            d += bf16_to_f32(q0[e]) * bf16_to_f32(kr[e]);
        if (d != 0.f) local++;
    }
    atomicAdd(&cnt, local);
    __syncthreads();
    if (threadIdx.x == 0) lengths[b] = cnt;
}

// ------------------------------ row softmax --------------------------------
__global__ __launch_bounds__(256)
void softmax_kernel(const float* __restrict__ S,
                    unsigned short* __restrict__ P) {
    const size_t row = blockIdx.x;
    const float* s = S + row * SEQ_T;
    unsigned short* p = P + row * SEQ_T;
    __shared__ float red[256];
    const float NEG_INF = -__builtin_inff();

    float m = NEG_INF;
    for (int j = threadIdx.x; j < SEQ_T; j += 256) m = fmaxf(m, s[j]);
    red[threadIdx.x] = m;
    __syncthreads();
    for (int off = 128; off > 0; off >>= 1) {
        if (threadIdx.x < off)
            red[threadIdx.x] = fmaxf(red[threadIdx.x], red[threadIdx.x + off]);
        __syncthreads();
    }
    m = red[0];
    __syncthreads();

    float sum = 0.f;
    for (int j = threadIdx.x; j < SEQ_T; j += 256) {
        float v = s[j];
        sum += (v == NEG_INF) ? 0.f : __expf(v - m);
    }
    red[threadIdx.x] = sum;
    __syncthreads();
    for (int off = 128; off > 0; off >>= 1) {
        if (threadIdx.x < off)
            red[threadIdx.x] += red[threadIdx.x + off];
        __syncthreads();
    }
    sum = red[0];
    float inv = (sum > 0.f) ? 1.f / sum : 0.f;

    for (int j = threadIdx.x; j < SEQ_T; j += 256) {
        float v = s[j];
        float e = (v == NEG_INF) ? 0.f : __expf(v - m) * inv;
        p[j] = f32_to_bf16(e);
    }
}

// ------------------------------- launcher ----------------------------------
extern "C" void kernel_launch(void* const* d_in, const int* in_sizes, int n_in,
                              void* d_out, int out_size, void* d_ws, size_t ws_size,
                              hipStream_t stream) {
    (void)in_sizes; (void)n_in; (void)out_size; (void)ws_size;
    const float* x  = (const float*)d_in[0];
    const float* Wq = (const float*)d_in[1];
    const float* Wk = (const float*)d_in[2];
    const float* Wv = (const float*)d_in[3];
    const int*   dk = (const int*)d_in[4];
    float* out = (float*)d_out;

    const size_t M  = (size_t)BATCH_N * SEQ_T;      // 16384
    const size_t nX = M * DIM_E;
    const size_t nW = (size_t)DIM_E * DIM_H;

    char* ws = (char*)d_ws;
    size_t off = 0;
    auto carve = [&](size_t bytes) -> char* {
        char* p = ws + off;
        off += (bytes + 255) & ~(size_t)255;
        return p;
    };
    unsigned short* xb  = (unsigned short*)carve(nX * 2);
    unsigned short* Wqt = (unsigned short*)carve(nW * 2);   // [H][E] bf16
    unsigned short* Wkt = (unsigned short*)carve(nW * 2);
    unsigned short* Wvt = (unsigned short*)carve(nW * 2);
    unsigned short* Qb  = (unsigned short*)carve(M * DIM_H * 2);   // [B*T][H]
    unsigned short* Kb  = (unsigned short*)carve(M * DIM_H * 2);   // [B*T][H]
    unsigned short* Vt  = (unsigned short*)carve(M * DIM_H * 2);   // [B][H][T]
    float*          S   = (float*)carve((size_t)BATCH_N * SEQ_T * SEQ_T * 4);
    unsigned short* P   = (unsigned short*)carve((size_t)BATCH_N * SEQ_T * SEQ_T * 2);
    int*            len = (int*)carve(BATCH_N * sizeof(int));

    // 1) operand down-convert (weights transposed -> everything becomes NT)
    cvt_f32_bf16_kernel<<<2048, 256, 0, stream>>>(x, xb, (long)(nX / 4));
    cvt_transpose_kernel<<<1024, 256, 0, stream>>>(Wq, Wqt);
    cvt_transpose_kernel<<<1024, 256, 0, stream>>>(Wk, Wkt);
    cvt_transpose_kernel<<<1024, 256, 0, stream>>>(Wv, Wvt);

    dim3 blk(256);

    // 2) projections: Q/K row-major bf16; V stored transposed per batch
    dim3 gproj(DIM_H / 128, (unsigned)(M / 128), 1);   // (8, 128)
    gemm_nt_kernel<EPI_BF16><<<gproj, blk, 0, stream>>>(
        xb, Wqt, Qb, (int)M, DIM_H, DIM_E, 0, 0, 0, nullptr, nullptr);
    gemm_nt_kernel<EPI_BF16><<<gproj, blk, 0, stream>>>(
        xb, Wkt, Kb, (int)M, DIM_H, DIM_E, 0, 0, 0, nullptr, nullptr);
    gemm_nt_kernel<EPI_BF16_T><<<gproj, blk, 0, stream>>>(
        xb, Wvt, Vt, (int)M, DIM_H, DIM_E, 0, 0, 0, nullptr, nullptr);

    // 3) padding-mask lengths from energy row 0
    lengths_kernel<<<BATCH_N, 256, 0, stream>>>(Qb, Kb, len);

    // 4) energy = scale * Q@K^T with causal+pad mask (fp32, L2-resident)
    dim3 ge(SEQ_T / 128, SEQ_T / 128, BATCH_N);        // (16, 16, 8)
    gemm_nt_kernel<EPI_ENERGY><<<ge, blk, 0, stream>>>(
        Qb, Kb, S, SEQ_T, SEQ_T, DIM_H,
        (long)SEQ_T * DIM_H, (long)SEQ_T * DIM_H, (long)SEQ_T * SEQ_T, len, dk);

    // 5) row softmax -> bf16 probabilities
    softmax_kernel<<<(unsigned)M, 256, 0, stream>>>(S, P);

    // 6) out = P @ V = P @ (Vt)^T  (fp32 out), batched
    dim3 gpv(DIM_H / 128, SEQ_T / 128, BATCH_N);       // (8, 16, 8)
    gemm_nt_kernel<EPI_F32><<<gpv, blk, 0, stream>>>(
        P, Vt, out, SEQ_T, DIM_H, SEQ_T,
        (long)SEQ_T * SEQ_T, (long)SEQ_T * DIM_H, (long)SEQ_T * DIM_H,
        nullptr, nullptr);
}
